// GCN_4681514352906
// MI455X (gfx1250) — compile-verified, matrix operand order checked by reading, not run on hardware
//
#include <hip/hip_runtime.h>
#include <hip/hip_bf16.h>
#include <cstdint>
#include <cstddef>

// ---------------------------------------------------------------------------
// GCN (2-layer) for MI455X / gfx1250, wave32, WMMA bf16 path.
//   h1 = norm-agg(x @ W1) + b1 ; relu ; h2 = norm-agg(relu(h1) @ W2) + b2
//   out = log_softmax(h2)
// GEMMs: one wave owns a 16-wide N panel of B in registers (bf16) and streams
// a strip of 16-row M tiles through v_wmma_f32_16x16x32_bf16 (f32 accum).
// Aggregation: float4 gather + global_atomic_add_f32 scatter (bandwidth-bound).
// ---------------------------------------------------------------------------

typedef __attribute__((ext_vector_type(16))) __bf16 v16bf;
typedef __attribute__((ext_vector_type(8)))  float  v8f;

#define KDIM 128   // inner dimension for BOTH gemms (F_in == H == 128)

// ---------------- degree / norm ----------------
__global__ void k_init_deg(float* __restrict__ deg, int n) {
    int i = blockIdx.x * blockDim.x + threadIdx.x;
    if (i < n) deg[i] = 1.0f;                       // self-loop contribution
}

__global__ void k_edge_deg(const int* __restrict__ dst, float* __restrict__ deg, int E) {
    int e = blockIdx.x * blockDim.x + threadIdx.x;
    if (e < E) atomicAdd(&deg[dst[e]], 1.0f);
}

__global__ void k_dinv(float* __restrict__ deg, int n) {
    int i = blockIdx.x * blockDim.x + threadIdx.x;
    if (i < n) deg[i] = rsqrtf(fmaxf(deg[i], 1.0f));
}

// ---------------- WMMA GEMM: C[M,N] = A[M,128] * B[128,N], optional relu(A) --
__device__ __forceinline__ float4 relu4(float4 v) {
    return make_float4(fmaxf(v.x, 0.f), fmaxf(v.y, 0.f),
                       fmaxf(v.z, 0.f), fmaxf(v.w, 0.f));
}

__global__ void k_gemm_wmma_bf16(const float* __restrict__ A,
                                 const float* __restrict__ B,
                                 float* __restrict__ C,
                                 int Mtiles, int N, int applyRelu, int mtPerWave) {
    const int wavesPerBlock = blockDim.x >> 5;
    const int w       = blockIdx.x * wavesPerBlock + (threadIdx.x >> 5);
    const int Ntiles  = N >> 4;
    const int nChunks = (Mtiles + mtPerWave - 1) / mtPerWave;
    if (w >= Ntiles * nChunks) return;              // wave-uniform: EXEC stays all-1s
    const int nt    = w % Ntiles;
    const int chunk = w / Ntiles;
    const int mt0   = chunk * mtPerWave;
    const int mt1   = (mt0 + mtPerWave < Mtiles) ? (mt0 + mtPerWave) : Mtiles;

    const int lane = threadIdx.x & 31;
    const int half = lane >> 4;                     // K-half selector (ISA A/B layout)
    const int lo16 = lane & 15;
    const int n    = (nt << 4) + lo16;              // B column owned by this lane

    // ---- load B panel once: 128x16 slice of B, bf16, register-resident ----
    // lane holds column n; per kk: K = [kb,kb+8) and [kb+16,kb+24), kb = kk*32+half*8
    const float* __restrict__ bcol = B + n;
    v16bf bmat[KDIM / 32];
#pragma unroll
    for (int kk = 0; kk < KDIM / 32; ++kk) {
        const int kb = kk * 32 + half * 8;
#pragma unroll
        for (int i = 0; i < 8; ++i) {
            bmat[kk][i]     = (__bf16)bcol[(size_t)(kb + i)      * N];
            bmat[kk][8 + i] = (__bf16)bcol[(size_t)(kb + 16 + i) * N];
        }
    }

    // ---- stream M tiles through the WMMA pipe ----
    for (int mt = mt0; mt < mt1; ++mt) {
        const float* __restrict__ arow = A + (size_t)((mt << 4) + lo16) * KDIM;
        // prefetch next tile's A row (lowered to global_prefetch_b8)
        __builtin_prefetch((const void*)(arow + ((size_t)KDIM << 4)), 0, 0);

        v8f acc = {};
#pragma unroll
        for (int kk = 0; kk < KDIM / 32; ++kk) {
            const int kb = kk * 32 + half * 8;
            float4 f0 = *(const float4*)(arow + kb);
            float4 f1 = *(const float4*)(arow + kb + 4);
            float4 f2 = *(const float4*)(arow + kb + 16);
            float4 f3 = *(const float4*)(arow + kb + 20);
            if (applyRelu) { f0 = relu4(f0); f1 = relu4(f1); f2 = relu4(f2); f3 = relu4(f3); }

            v16bf a;
            a[0]  = (__bf16)f0.x; a[1]  = (__bf16)f0.y; a[2]  = (__bf16)f0.z; a[3]  = (__bf16)f0.w;
            a[4]  = (__bf16)f1.x; a[5]  = (__bf16)f1.y; a[6]  = (__bf16)f1.z; a[7]  = (__bf16)f1.w;
            a[8]  = (__bf16)f2.x; a[9]  = (__bf16)f2.y; a[10] = (__bf16)f2.z; a[11] = (__bf16)f2.w;
            a[12] = (__bf16)f3.x; a[13] = (__bf16)f3.y; a[14] = (__bf16)f3.z; a[15] = (__bf16)f3.w;

            acc = __builtin_amdgcn_wmma_f32_16x16x32_bf16(
                      /*neg_a=*/false, a, /*neg_b=*/false, bmat[kk],
                      /*c_mod=*/(short)0, acc, /*reuse_a=*/false, /*reuse_b=*/false);
        }

        // C/D layout: VGPR r -> row (mt*16 + r + 8*half), N = lo16 column
        float* __restrict__ crow = C + (size_t)((mt << 4) + 8 * half) * N + n;
#pragma unroll
        for (int r = 0; r < 8; ++r)
            crow[(size_t)r * N] = acc[r];
    }
}

// ---------------- aggregation ----------------
// AGG[i, 4q..4q+3] = bias + dinv[i]^2 * Hmat[i]   (self-loop term + init)
__global__ void k_self_agg4(const float* __restrict__ Hmat,
                            const float* __restrict__ dinv,
                            const float* __restrict__ bias,
                            float* __restrict__ AGG, int n, int Fq /* F/4 */) {
    int idx = blockIdx.x * blockDim.x + threadIdx.x;
    if (idx >= n * Fq) return;
    int i = idx / Fq;
    int q = idx - i * Fq;
    float d  = dinv[i];
    float dd = d * d;
    float4 h  = ((const float4*)Hmat)[idx];
    float4 bv = ((const float4*)bias)[q];
    ((float4*)AGG)[idx] = make_float4(bv.x + dd * h.x, bv.y + dd * h.y,
                                      bv.z + dd * h.z, bv.w + dd * h.w);
}

// AGG[dst, 4q..4q+3] += dinv[src]*dinv[dst] * Hmat[src, 4q..4q+3] per edge
__global__ void k_edge_scatter4(const int* __restrict__ src,
                                const int* __restrict__ dst,
                                const float* __restrict__ Hmat,
                                const float* __restrict__ dinv,
                                float* __restrict__ AGG,
                                int E, int logFq, int F) {
    long long idx = (long long)blockIdx.x * blockDim.x + threadIdx.x;
    if (idx >= ((long long)E << logFq)) return;
    int e = (int)(idx >> logFq);
    int q = (int)(idx & ((1 << logFq) - 1));
    int s = src[e];
    int d = dst[e];
    float norm = dinv[s] * dinv[d];
    float4 h = *(const float4*)(Hmat + (size_t)s * F + (q << 2));
    float* out = AGG + (size_t)d * F + (q << 2);
    atomicAdd(out + 0, norm * h.x);
    atomicAdd(out + 1, norm * h.y);
    atomicAdd(out + 2, norm * h.z);
    atomicAdd(out + 3, norm * h.w);
}

// ---------------- log-softmax over C classes per node ----------------
__global__ void k_log_softmax(const float* __restrict__ AGG,
                              float* __restrict__ out, int n, int C) {
    int i = blockIdx.x * blockDim.x + threadIdx.x;
    if (i >= n) return;
    const float* row = AGG + (size_t)i * C;
    float mx = -3.402823466e38f;
    for (int c = 0; c < C; ++c) mx = fmaxf(mx, row[c]);
    float s = 0.f;
    for (int c = 0; c < C; ++c) s += __expf(row[c] - mx);
    float lse = mx + __logf(s);
    float* orow = out + (size_t)i * C;
    for (int c = 0; c < C; ++c) orow[c] = row[c] - lse;
}

// ---------------------------------------------------------------------------
extern "C" void kernel_launch(void* const* d_in, const int* in_sizes, int n_in,
                              void* d_out, int out_size, void* d_ws, size_t ws_size,
                              hipStream_t stream) {
    const float* x  = (const float*)d_in[0];
    const int*   ei = (const int*)  d_in[1];
    const float* W1 = (const float*)d_in[2];
    const float* b1 = (const float*)d_in[3];
    const float* W2 = (const float*)d_in[4];
    const float* b2 = (const float*)d_in[5];

    const int N = in_sizes[0] / 128;   // 50000 nodes
    const int E = in_sizes[1] / 2;     // 800000 edges
    const int H = 128;
    const int C = in_sizes[5];         // 16 classes

    const int* src = ei;
    const int* dst = ei + E;

    // workspace carve-out (256B aligned slices)
    char*  ws  = (char*)d_ws;
    size_t off = 0;
    auto carve = [&](size_t bytes) {
        char* p = ws + off;
        off = (off + bytes + 255) & ~(size_t)255;
        return p;
    };
    float* dinv = (float*)carve((size_t)N * 4);          // deg -> dinv in place
    float* H1   = (float*)carve((size_t)N * H * 4);
    float* A1   = (float*)carve((size_t)N * H * 4);
    float* H2   = (float*)carve((size_t)N * C * 4);
    float* A2   = (float*)carve((size_t)N * C * 4);
    float* out  = (float*)d_out;

    const int T = 256;

    // 1) degrees + symmetric norm
    k_init_deg<<<(N + T - 1) / T, T, 0, stream>>>(dinv, N);
    k_edge_deg<<<(E + T - 1) / T, T, 0, stream>>>(dst, dinv, E);
    k_dinv    <<<(N + T - 1) / T, T, 0, stream>>>(dinv, N);

    const int Mtiles = N / 16;         // 3125 (N divisible by 16)

    // 2) H1 = x @ W1   (WMMA bf16, f32 accumulate; B panel register-resident)
    {
        int Ntiles  = H / 16;                       // 8
        int mtPW    = 8;                            // M tiles per wave
        int nChunks = (Mtiles + mtPW - 1) / mtPW;
        int waves   = Ntiles * nChunks;
        k_gemm_wmma_bf16<<<(waves + 3) / 4, 128, 0, stream>>>(
            x, W1, H1, Mtiles, H, /*relu=*/0, mtPW);
    }

    // 3) AGG1 = b1 + D^-1/2 (A+I) D^-1/2 H1
    k_self_agg4<<<((N * (H / 4)) + T - 1) / T, T, 0, stream>>>(H1, dinv, b1, A1, N, H / 4);
    {
        long long tot = (long long)E * (H / 4);     // one thread per 4 features
        k_edge_scatter4<<<(int)((tot + T - 1) / T), T, 0, stream>>>(
            src, dst, H1, dinv, A1, E, /*logFq=*/5, H);
    }

    // 4) H2 = relu(AGG1) @ W2   (relu fused into A-tile conversion)
    {
        int Ntiles  = C / 16;                       // 1
        int mtPW    = 8;
        int nChunks = (Mtiles + mtPW - 1) / mtPW;
        int waves   = Ntiles * nChunks;
        k_gemm_wmma_bf16<<<(waves + 3) / 4, 128, 0, stream>>>(
            A1, W2, H2, Mtiles, C, /*relu=*/1, mtPW);
    }

    // 5) AGG2 = b2 + D^-1/2 (A+I) D^-1/2 H2
    k_self_agg4<<<((N * (C / 4)) + T - 1) / T, T, 0, stream>>>(H2, dinv, b2, A2, N, C / 4);
    {
        long long tot = (long long)E * (C / 4);
        k_edge_scatter4<<<(int)((tot + T - 1) / T), T, 0, stream>>>(
            src, dst, H2, dinv, A2, E, /*logFq=*/2, C);
    }

    // 6) out = log_softmax(AGG2)
    k_log_softmax<<<(N + T - 1) / T, T, 0, stream>>>(A2, out, N, C);
}
